// Self_Attention_70626442215917
// MI455X (gfx1250) — matrix operand order repaired and checked by reading, not compiled, not run on hardware
//
#include <hip/hip_runtime.h>
#include <hip/hip_bf16.h>
#include <stdint.h>

typedef __bf16 bf16_t;
typedef __attribute__((ext_vector_type(16))) __bf16 v16bf;
typedef __attribute__((ext_vector_type(8)))  __bf16 v8bf;
typedef __attribute__((ext_vector_type(8)))  float  v8f;

#define B_ 8
#define S_ 2048
#define D_ 512
#define MT 32   // queries per block (2 row-groups of 16)
#define KT 64   // keys per tile (4 waves x 16 cols)

__device__ __forceinline__ float rmax16(float v) {
#pragma unroll
  for (int m = 1; m < 16; m <<= 1) v = fmaxf(v, __shfl_xor(v, m, 32));
  return v;
}
__device__ __forceinline__ float rsum16(float v) {
#pragma unroll
  for (int m = 1; m < 16; m <<= 1) v += __shfl_xor(v, m, 32);
  return v;
}

// Pass 1: qk = bf16(tanh(x)) row-major [B][S][D]; vT = bf16(x) transposed [B][D][S].
__global__ __launch_bounds__(256, 1) void prep_kernel(const float* __restrict__ x,
                                                      bf16_t* __restrict__ qk,
                                                      bf16_t* __restrict__ vT) {
  __shared__ float tile[32][33];
  const int b   = blockIdx.z;
  const int t0  = blockIdx.y * 32;
  const int d0  = blockIdx.x * 32;
  const int col = threadIdx.x & 31;
  const int row = threadIdx.x >> 5;  // 8 rows per pass
#pragma unroll
  for (int rr = 0; rr < 32; rr += 8) {
    int t = t0 + row + rr;
    float v = x[((size_t)b * S_ + t) * D_ + d0 + col];
    tile[row + rr][col] = v;
    qk[((size_t)b * S_ + t) * D_ + d0 + col] = (bf16_t)tanhf(v);
  }
  __syncthreads();
#pragma unroll
  for (int rr = 0; rr < 32; rr += 8) {
    int d = d0 + row + rr;
    vT[((size_t)b * D_ + d) * S_ + t0 + col] = (bf16_t)tile[col][row + rr];
  }
}

// Pass 2: flash attention. 8 waves: wave = (r<<2)|c ; r = 16-query row group,
// c = 16-key column slice for scores AND 128-wide D slice for the PV stage.
__global__ __launch_bounds__(256, 1) void attn_kernel(const bf16_t* __restrict__ qk,
                                                      const bf16_t* __restrict__ vT,
                                                      float* __restrict__ out) {
  __shared__ __align__(16) bf16_t Qs[MT][D_];       // 32 KB
  __shared__ __align__(16) bf16_t Ps[MT][KT + 8];   // padded: 144B rows, 16B aligned
  __shared__ float partmax[2][4][16];
  __shared__ float partsum[2][4][16];
  __shared__ float mrow[MT];
  __shared__ float lrow[MT];

  const int b    = blockIdx.y;
  const int q0   = blockIdx.x * MT;
  const int tid  = threadIdx.x;
  const int lane = tid & 31;
  const int wv   = tid >> 5;
  const int r    = wv >> 2;
  const int c    = wv & 3;
  const int l15  = lane & 15;
  const int hi   = lane >> 4;

  // Stage Q tile (32x512 bf16) into LDS with async LDS-DMA (ASYNCcnt path).
  const bf16_t* qsrc = qk + ((size_t)b * S_ + q0) * D_;
#pragma unroll
  for (int i = 0; i < (MT * D_) / (256 * 8); ++i) {
    int off = (tid + i * 256) * 8;                       // 8 bf16 = 16 B per lane
    unsigned lds_off = (unsigned)(uintptr_t)(&Qs[0][0] + off);
    const bf16_t* g = qsrc + off;
    asm volatile("global_load_async_to_lds_b128 %0, %1, off"
                 :: "v"(lds_off), "v"(g) : "memory");
  }
  asm volatile("s_wait_asynccnt 0x0" ::: "memory");
  if (tid < MT) { mrow[tid] = -__builtin_inff(); lrow[tid] = 0.0f; }
  __syncthreads();

  const float scale = 0.04419417382415922f;  // 1/sqrt(512)
  const int arow   = r * 16 + l15;  // A-operand row for this lane
  const int khalf  = hi * 16;       // A-operand K sub-offset (contiguous k = 16h + s)
  const int kbhalf = hi * 16;       // B-operand K sub-offset

  v8f oacc[8] = {};  // 16 x 128 f32 output accumulator (8 C/D tiles)

  for (int jt = 0; jt < S_ / KT; ++jt) {
    // ---- scores: S_rc[16x16] = Q_r(16xD) . K_c(16xD)^T, K streamed from L2 ----
    const bf16_t* kb = qk + ((size_t)b * S_ + jt * KT + c * 16 + l15) * D_ + kbhalf;
    if (jt + 1 < S_ / KT)
      __builtin_prefetch(qk + ((size_t)b * S_ + (jt + 1) * KT + c * 16 + l15) * D_, 0, 3);

    v8f acc = {};
#pragma unroll
    for (int kk = 0; kk < D_ / 32; ++kk) {
      union { v16bf v; v8bf h[2]; } a;
      a.h[0] = *(const v8bf*)&Qs[arow][kk * 32 + khalf];
      a.h[1] = *(const v8bf*)&Qs[arow][kk * 32 + 8 + khalf];
      v16bf bm = *(const v16bf*)(kb + kk * 32);
      acc = __builtin_amdgcn_wmma_f32_16x16x32_bf16(false, a.v, false, bm,
                                                    (short)0, acc, false, false);
    }

    // ---- online softmax across the 64-key tile ----
    float pmax[8];
#pragma unroll
    for (int g = 0; g < 8; ++g) { acc[g] *= scale; pmax[g] = rmax16(acc[g]); }
    if (l15 == 0) {
#pragma unroll
      for (int g = 0; g < 8; ++g) partmax[r][c][g + hi * 8] = pmax[g];
    }
    __syncthreads();  // partial maxima visible

    float mnew[8], alpha[8];
#pragma unroll
    for (int g = 0; g < 8; ++g) {
      int row = g + hi * 8;
      float tm = fmaxf(fmaxf(partmax[r][0][row], partmax[r][1][row]),
                       fmaxf(partmax[r][2][row], partmax[r][3][row]));
      float mo = mrow[r * 16 + row];
      float mn = fmaxf(mo, tm);
      mnew[g]  = mn;
      alpha[g] = __expf(mo - mn);
    }
    float psum[8];
#pragma unroll
    for (int g = 0; g < 8; ++g) {
      float p = __expf(acc[g] - mnew[g]);
      psum[g] = rsum16(p);
      Ps[r * 16 + g + hi * 8][c * 16 + l15] = (bf16_t)p;
    }
    if (l15 == 0) {
#pragma unroll
      for (int g = 0; g < 8; ++g) partsum[r][c][g + hi * 8] = psum[g];
    }
#pragma unroll
    for (int t = 0; t < 8; ++t)
#pragma unroll
      for (int g = 0; g < 8; ++g) oacc[t][g] *= alpha[g];
    __syncthreads();  // P + partial sums visible

    if (c == 0 && lane < 16) {  // one wave per row-group updates running stats
      int row = r * 16 + lane;
      float s  = partsum[r][0][lane] + partsum[r][1][lane] +
                 partsum[r][2][lane] + partsum[r][3][lane];
      float tm = fmaxf(fmaxf(partmax[r][0][lane], partmax[r][1][lane]),
                       fmaxf(partmax[r][2][lane], partmax[r][3][lane]));
      float mo = mrow[row];
      float mn = fmaxf(mo, tm);
      lrow[row] = lrow[row] * __expf(mo - mn) + s;
      mrow[row] = mn;
    }

    // ---- O_rc(16x128) += P_r(16x64) . V(64x128), V^T streamed from L2 ----
    // kk outer + batched B preloads so the 8-WMMA chain pipelines like the
    // score stage (clause of loads, staggered s_wait_loadcnt) instead of
    // load->wait(0)->wmma serialization.
#pragma unroll
    for (int kk = 0; kk < KT / 32; ++kk) {
      v16bf bmv[8];
#pragma unroll
      for (int t = 0; t < 8; ++t) {
        int dcol = c * 128 + t * 16 + l15;
        const bf16_t* vb = vT + ((size_t)b * D_ + dcol) * S_ + jt * KT + kbhalf;
        bmv[t] = *(const v16bf*)(vb + kk * 32);
      }
      union { v16bf v; v8bf h[2]; } a;
      a.h[0] = *(const v8bf*)&Ps[arow][kk * 32 + khalf];
      a.h[1] = *(const v8bf*)&Ps[arow][kk * 32 + 8 + khalf];
#pragma unroll
      for (int t = 0; t < 8; ++t) {
        oacc[t] = __builtin_amdgcn_wmma_f32_16x16x32_bf16(false, a.v, false, bmv[t],
                                                          (short)0, oacc[t], false, false);
      }
    }
  }

  __syncthreads();  // final lrow visible
  float rcl[8];
#pragma unroll
  for (int g = 0; g < 8; ++g) rcl[g] = 1.0f / lrow[r * 16 + g + hi * 8];
#pragma unroll
  for (int t = 0; t < 8; ++t) {
#pragma unroll
    for (int g = 0; g < 8; ++g) {
      int m = q0 + r * 16 + g + hi * 8;
      int d = c * 128 + t * 16 + l15;
      out[((size_t)b * S_ + m) * D_ + d] = oacc[t][g] * rcl[g];
    }
  }
}

extern "C" void kernel_launch(void* const* d_in, const int* in_sizes, int n_in,
                              void* d_out, int out_size, void* d_ws, size_t ws_size,
                              hipStream_t stream) {
  (void)in_sizes; (void)n_in; (void)out_size; (void)ws_size;
  const float* x = (const float*)d_in[0];
  float* out = (float*)d_out;
  bf16_t* qk = (bf16_t*)d_ws;                       // 16 MB
  bf16_t* vT = qk + (size_t)B_ * S_ * D_;           // 16 MB

  dim3 gp(D_ / 32, S_ / 32, B_);
  prep_kernel<<<gp, 256, 0, stream>>>(x, qk, vT);

  dim3 ga(S_ / MT, B_);
  attn_kernel<<<ga, 256, 0, stream>>>(qk, vT, out);
}
